// GuidedUpCatConv1x1_72782515798655
// MI455X (gfx1250) — matrix operand order repaired and checked by reading, not compiled, not run on hardware
//
#include <hip/hip_runtime.h>
#include <hip/hip_bf16.h>
#include <math.h>

// ---------------------------------------------------------------------------
// Problem constants (from reference)
// ---------------------------------------------------------------------------
#define BB   4
#define CIN  256
#define GC   64
#define PL   64
#define CCAT 320      // CIN + GC
#define HL   128
#define WL   128
#define HH   256
#define WW   256
#define NPIX (BB * HH * WW)          // 262144 pixels total
#define BN_EPS 1e-5f

typedef __bf16 bf16_t;
typedef bf16_t v16bf __attribute__((ext_vector_type(16)));
typedef float  v8f   __attribute__((ext_vector_type(8)));
typedef unsigned int u32;
typedef u32 v4u __attribute__((ext_vector_type(4)));

static __device__ __forceinline__ u32 pack2_bf16(float lo, float hi) {
    const unsigned short a = __builtin_bit_cast(unsigned short, (bf16_t)lo);
    const unsigned short b = __builtin_bit_cast(unsigned short, (bf16_t)hi);
    return (u32)a | ((u32)b << 16);
}

// ---------------------------------------------------------------------------
// Pass 0: zero the BN accumulators (sum[64], sumsq[64])
// ---------------------------------------------------------------------------
__global__ void zero_stats_kernel(float* __restrict__ stats) {
    if (threadIdx.x < 128) stats[threadIdx.x] = 0.0f;
}

// ---------------------------------------------------------------------------
// Pass 1: fused bilinear-upsample + concat + 1x1 conv as a WMMA GEMM.
//   xv[b,o,h,w] = sum_c W[o,c] * cat[b,c,h,w]
// Block: 128 threads (4 waves), 64 consecutive pixels of one row per block.
// Both operands are staged in LDS as bf16 pair-dwords, pre-arranged in the
// exact WMMA lane layout, so the inner loop is pure ds_load_b128 + v_wmma.
//   A frag (lane): dwords [base..base+3] and [base+8..base+11],
//                  base = (m0+l15)*PKROW + kt*16 + lhalf*4
//   B frag (lane): dwords [b0..b0+3] and [b0+8..b0+11], b0 = p*PKROW + kt*16 + lhalf*4
// ---------------------------------------------------------------------------
#define KPAIRS 160    // 320 channels / 2 per dword
#define PKROW  164    // 160 + 4 pad dwords: 16B-aligned rows, bank stride 36

__global__ __launch_bounds__(128) void conv1x1_wmma_kernel(
    const float* __restrict__ x, const float* __restrict__ guide,
    const float* __restrict__ Wc, float* __restrict__ xv)
{
    __shared__ __align__(16) u32 w_pk[PL * PKROW];    // 42 KB
    __shared__ __align__(16) u32 cat_pk[64 * PKROW];  // 42 KB

    const int tid = threadIdx.x;
    const int seg = blockIdx.x;                  // 4096 segments
    const int w0  = (seg & 3) * 64;
    const int h   = (seg >> 2) & (HH - 1);
    const int b   = seg >> 10;

    // --- stage weights as packed bf16 pairs -------------------------------
    for (int i = tid; i < PL * KPAIRS; i += 128) {
        const int o = i / KPAIRS, kp = i - o * KPAIRS;
        const float* wr = Wc + o * CCAT + 2 * kp;
        w_pk[o * PKROW + kp] = pack2_bf16(wr[0], wr[1]);
    }

    // --- build cat tile: 160 channel-pairs x 64 pixels --------------------
    {
        const int p    = tid & 63;               // fixed pixel per thread
        const int wpix = w0 + p;
        // jax.image.resize 'bilinear': src = (dst+0.5)*0.5 - 0.5
        int i0r = (h - 1) >> 1;
        const float fh = (h & 1) ? 0.25f : 0.75f;
        int i1 = i0r + 1; if (i1 > HL - 1) i1 = HL - 1;
        int i0 = i0r < 0 ? 0 : i0r;
        int j0r = (wpix - 1) >> 1;
        const float fw = (wpix & 1) ? 0.25f : 0.75f;
        int j1 = j0r + 1; if (j1 > WL - 1) j1 = WL - 1;
        int j0 = j0r < 0 ? 0 : j0r;
        const float w00 = (1.f - fh) * (1.f - fw), w01 = (1.f - fh) * fw;
        const float w10 = fh * (1.f - fw),         w11 = fh * fw;

        const int cp0 = tid >> 6;                // 0 or 1, stride 2 over pairs
        for (int cp = cp0; cp < 128; cp += 2) {  // channels 2cp, 2cp+1 < 256
            const float* xb0 = x + (((size_t)b * CIN + 2 * cp) * HL) * WL;
            const float* xb1 = xb0 + (size_t)HL * WL;
            const float v0 = w00 * xb0[i0 * WL + j0] + w01 * xb0[i0 * WL + j1]
                           + w10 * xb0[i1 * WL + j0] + w11 * xb0[i1 * WL + j1];
            const float v1 = w00 * xb1[i0 * WL + j0] + w01 * xb1[i0 * WL + j1]
                           + w10 * xb1[i1 * WL + j0] + w11 * xb1[i1 * WL + j1];
            cat_pk[p * PKROW + cp] = pack2_bf16(v0, v1);
        }
        for (int cp = 128 + cp0; cp < KPAIRS; cp += 2) {  // guide channels
            const float* gb = guide + ((((size_t)b * GC + (2 * cp - 256)) * HH + h) * WW) + wpix;
            cat_pk[p * PKROW + cp] = pack2_bf16(gb[0], gb[(size_t)HH * WW]);
        }
    }
    __syncthreads();

    // --- WMMA GEMM --------------------------------------------------------
    const int lane   = tid & 31;
    const int lhalf  = lane >> 4;                // 0: lanes 0-15, 1: 16-31
    const int l15    = lane & 15;
    const int m0     = (tid >> 5) * 16;          // wave -> M-tile

    v8f acc[4];
    #pragma unroll
    for (int nt = 0; nt < 4; ++nt) acc[nt] = (v8f){0,0,0,0,0,0,0,0};

    const int arow = (m0 + l15) * PKROW + lhalf * 4;

    #pragma unroll 2
    for (int kt = 0; kt < 10; ++kt) {
        union { v4u q[2]; v16bf bf; } ua;
        ua.q[0] = *(const v4u*)&w_pk[arow + kt * 16];
        ua.q[1] = *(const v4u*)&w_pk[arow + kt * 16 + 8];
        const v16bf a = ua.bf;
        #pragma unroll
        for (int nt = 0; nt < 4; ++nt) {
            const int b0 = (nt * 16 + l15) * PKROW + kt * 16 + lhalf * 4;
            union { v4u q[2]; v16bf bf; } ub;
            ub.q[0] = *(const v4u*)&cat_pk[b0];
            ub.q[1] = *(const v4u*)&cat_pk[b0 + 8];
            acc[nt] = __builtin_amdgcn_wmma_f32_16x16x32_bf16(
                false, a, false, ub.bf, (short)0, acc[nt], false, false);
        }
    }

    // --- store D tiles: VGPR r -> M = r + 8*lhalf, N = l15 ----------------
    #pragma unroll
    for (int nt = 0; nt < 4; ++nt) {
        const int pcol = nt * 16 + l15;
        #pragma unroll
        for (int r = 0; r < 8; ++r) {
            const int o = m0 + r + lhalf * 8;
            xv[(((size_t)b * PL + o) * HH + h) * WW + w0 + pcol] = acc[nt][r];
        }
    }
}

// ---------------------------------------------------------------------------
// Pass 2: 3x3 local attention (q=k=guide, v=xv) + BN statistics.
// Block = 256 threads = one 16x16 pixel tile; channels processed in two
// 32-channel chunks staged in LDS with a 1-pixel zero halo (18x18).
// ---------------------------------------------------------------------------
#define TSTRIDE 324   // 18*18
#define OSTRIDE 257   // 256 + 1 pad

__global__ __launch_bounds__(256) void attn_kernel(
    const float* __restrict__ guide, const float* __restrict__ xv,
    float* __restrict__ out, float* __restrict__ stats)
{
    __shared__ float stage[32 * TSTRIDE];    // 41.5 KB  (guide, then xv)
    __shared__ float out_lds[32 * OSTRIDE];  // 32.9 KB
    __shared__ float red[256];               // 32 ch * 4 parts * {sum,sumsq}

    const int tid = threadIdx.x;
    const int tx = blockIdx.x & 15, ty = (blockIdx.x >> 4) & 15, b = blockIdx.x >> 8;
    const int h0 = ty * 16, w0 = tx * 16;
    const int py = tid >> 4, px = tid & 15;

    float s[9];
    #pragma unroll
    for (int t = 0; t < 9; ++t) s[t] = 0.0f;

    // ---- attention scores over both channel chunks -----------------------
    for (int chunk = 0; chunk < 2; ++chunk) {
        for (int i = tid; i < 32 * TSTRIDE; i += 256) {
            const int c = i / TSTRIDE, r = i - c * TSTRIDE;
            const int dy = r / 18, dx = r - dy * 18;
            const int hh = h0 + dy - 1, ww2 = w0 + dx - 1;
            float v = 0.0f;
            if (hh >= 0 && hh < HH && ww2 >= 0 && ww2 < WW)
                v = guide[(((size_t)b * GC + chunk * 32 + c) * HH + hh) * WW + ww2];
            stage[i] = v;
        }
        __syncthreads();
        #pragma unroll 4
        for (int c = 0; c < 32; ++c) {
            const float* g = &stage[c * TSTRIDE];
            const float q = g[(py + 1) * 18 + (px + 1)];
            #pragma unroll
            for (int t = 0; t < 9; ++t)
                s[t] += q * g[(py + t / 3) * 18 + (px + t % 3)];
        }
        __syncthreads();
    }

    // ---- softmax over the 9 taps (scale 1/sqrt(64)) ----------------------
    #pragma unroll
    for (int t = 0; t < 9; ++t) s[t] *= 0.125f;
    float mx = s[0];
    #pragma unroll
    for (int t = 1; t < 9; ++t) mx = fmaxf(mx, s[t]);
    float sum = 0.0f;
    #pragma unroll
    for (int t = 0; t < 9; ++t) { s[t] = __expf(s[t] - mx); sum += s[t]; }
    const float inv = 1.0f / sum;
    #pragma unroll
    for (int t = 0; t < 9; ++t) s[t] *= inv;

    // ---- weighted sum of v + BN partial sums -----------------------------
    const int hh = h0 + py, wcol = w0 + px;
    for (int chunk = 0; chunk < 2; ++chunk) {
        for (int i = tid; i < 32 * TSTRIDE; i += 256) {
            const int c = i / TSTRIDE, r = i - c * TSTRIDE;
            const int dy = r / 18, dx = r - dy * 18;
            const int h2 = h0 + dy - 1, w2 = w0 + dx - 1;
            float v = 0.0f;
            if (h2 >= 0 && h2 < HH && w2 >= 0 && w2 < WW)
                v = xv[(((size_t)b * PL + chunk * 32 + c) * HH + h2) * WW + w2];
            stage[i] = v;
        }
        __syncthreads();
        #pragma unroll 2
        for (int c = 0; c < 32; ++c) {
            const float* vv = &stage[c * TSTRIDE];
            float o = 0.0f;
            #pragma unroll
            for (int t = 0; t < 9; ++t)
                o += s[t] * vv[(py + t / 3) * 18 + (px + t % 3)];
            out[(((size_t)b * PL + chunk * 32 + c) * HH + hh) * WW + wcol] = o;
            out_lds[c * OSTRIDE + py * 16 + px] = o;
        }
        __syncthreads();
        if (tid < 128) {                       // 4 partial reducers / channel
            const int c = tid >> 2, part = tid & 3;
            const float* op = &out_lds[c * OSTRIDE + part * 64];
            float s1 = 0.0f, s2 = 0.0f;
            #pragma unroll 8
            for (int i = 0; i < 64; ++i) { const float v = op[i]; s1 += v; s2 += v * v; }
            red[c * 4 + part] = s1;
            red[128 + c * 4 + part] = s2;
        }
        __syncthreads();
        if (tid < 32) {
            const float s1 = red[tid * 4] + red[tid * 4 + 1] + red[tid * 4 + 2] + red[tid * 4 + 3];
            const float s2 = red[128 + tid * 4] + red[128 + tid * 4 + 1]
                           + red[128 + tid * 4 + 2] + red[128 + tid * 4 + 3];
            atomicAdd(&stats[chunk * 32 + tid], s1);
            atomicAdd(&stats[64 + chunk * 32 + tid], s2);
        }
        __syncthreads();
    }
}

// ---------------------------------------------------------------------------
// Pass 3: finalize BN -> per-channel scale/bias
// ---------------------------------------------------------------------------
__global__ void bn_finalize_kernel(const float* __restrict__ stats,
                                   const float* __restrict__ gamma,
                                   const float* __restrict__ beta,
                                   float* __restrict__ sb)
{
    const int c = threadIdx.x;
    if (c >= PL) return;
    const float invN = 1.0f / (float)NPIX;
    const float mean = stats[c] * invN;
    const float var  = stats[64 + c] * invN - mean * mean;
    const float scale = gamma[c] * rsqrtf(var + BN_EPS);
    sb[c]      = scale;
    sb[64 + c] = beta[c] - mean * scale;
}

// ---------------------------------------------------------------------------
// Pass 4: apply BN in place on d_out (float4 vectorized)
// ---------------------------------------------------------------------------
__global__ __launch_bounds__(256) void bn_apply_kernel(float* __restrict__ out,
                                                       const float* __restrict__ sb)
{
    float4* o4 = (float4*)out;
    const size_t total4 = (size_t)BB * PL * HH * WW / 4;   // 4,194,304
    for (size_t idx = (size_t)blockIdx.x * 256 + threadIdx.x;
         idx < total4; idx += (size_t)gridDim.x * 256) {
        const int c = (int)((idx >> 14) & 63);   // 16384 float4 per (b,c) plane
        const float sc = sb[c], bi = sb[64 + c];
        float4 v = o4[idx];
        v.x = v.x * sc + bi; v.y = v.y * sc + bi;
        v.z = v.z * sc + bi; v.w = v.w * sc + bi;
        o4[idx] = v;
    }
}

// ---------------------------------------------------------------------------
// Launch
// ---------------------------------------------------------------------------
extern "C" void kernel_launch(void* const* d_in, const int* in_sizes, int n_in,
                              void* d_out, int out_size, void* d_ws, size_t ws_size,
                              hipStream_t stream)
{
    const float* x     = (const float*)d_in[0];
    const float* guide = (const float*)d_in[1];
    const float* Wc    = (const float*)d_in[2];
    const float* gamma = (const float*)d_in[3];
    const float* beta  = (const float*)d_in[4];
    float* out = (float*)d_out;

    // workspace layout: xv (64 MB) | stats (128 f32) | scale/bias (128 f32)
    float* xv    = (float*)d_ws;
    float* stats = xv + (size_t)BB * PL * HH * WW;
    float* sb    = stats + 128;

    zero_stats_kernel<<<1, 128, 0, stream>>>(stats);
    conv1x1_wmma_kernel<<<BB * HH * (WW / 64), 128, 0, stream>>>(x, guide, Wc, xv);
    attn_kernel<<<BB * (HH / 16) * (WW / 16), 256, 0, stream>>>(guide, xv, out, stats);
    bn_finalize_kernel<<<1, 64, 0, stream>>>(stats, gamma, beta, sb);
    bn_apply_kernel<<<4096, 256, 0, stream>>>(out, sb);
}
